// DOAM_35716948034518
// MI455X (gfx1250) — matrix operand order repaired and checked by reading, not compiled, not run on hardware
//
#include <hip/hip_runtime.h>
#include <hip/hip_bf16.h>
#include <stdint.h>

typedef __bf16 bf16_t;
typedef __attribute__((ext_vector_type(16))) __bf16 v16bf;
typedef __attribute__((ext_vector_type(8)))  __bf16 v8bf;
typedef __attribute__((ext_vector_type(8)))  float  v8f;

#define HW_ 300
#define BATCH_ 32
#define PX_PER_BLK 64
#define NTHREADS 128
#define KPAD_MAX 288
#define CIN_MAX 32
#define IN_W 68   // padded LDS row stride (66 columns used)

__device__ __forceinline__ bf16_t f2bf(float f) {
  uint32_t u = __builtin_bit_cast(uint32_t, f);
  uint32_t r = u + 0x7FFFu + ((u >> 16) & 1u);   // round-to-nearest-even
  uint16_t h = (uint16_t)(r >> 16);
  return __builtin_bit_cast(bf16_t, h);
}

// ---------------------------------------------------------------------------
// Implicit-GEMM 3x3 conv (padding=1) via V_WMMA_F32_16X16X32_BF16.
// M = output channels (16 per mtile), K = 9*(cin0+cin1) padded to 32,
// N = 64 pixels of one output row per block (4 waves x 16 pixels).
// Optional second input source in1 (nearest-upsampled pooled tensor, factor `up`)
// occupies channels [cin0, cin0+cin1). Optional second weight set (gated conv):
// rows [cout0, cout0+cout1) come from w1/bias1; gated epilogue stores
// g * sigmoid(mask) over cout0 channels.
// ---------------------------------------------------------------------------
__global__ __launch_bounds__(NTHREADS)
void DOAM_conv3x3_wmma(const float* __restrict__ in0, int cin0,
                       const float* __restrict__ in1, int cin1, int up,
                       const float* __restrict__ w0, const float* __restrict__ bias0, int cout0,
                       const float* __restrict__ w1, const float* __restrict__ bias1, int cout1,
                       float* __restrict__ out, int outCstride, int chBase, int gated)
{
  __shared__ __align__(16) bf16_t sA[16 * KPAD_MAX];          //  9,216 B
  __shared__ __align__(16) bf16_t sIn[CIN_MAX * 3 * IN_W];    // 13,056 B
  __shared__ __align__(16) bf16_t sB[PX_PER_BLK * KPAD_MAX];  // 36,864 B
  __shared__ float sEpi[16 * PX_PER_BLK];                     //  4,096 B

  const int tid   = threadIdx.x;
  const int cinT  = cin0 + cin1;
  const int K     = 9 * cinT;
  const int Kpad  = (K + 31) & ~31;
  const int coutT = cout0 + cout1;
  const int mtile = blockIdx.y;
  const int c0    = blockIdx.x * PX_PER_BLK;
  const int bz    = blockIdx.z;
  const int b     = bz / HW_;
  const int r     = bz % HW_;

  // ---- prefetch halo row for the *next* output row (r+1 needs row r+2) ----
  // cacheline-granular (16 floats); pulls toward L2 ahead of that block's
  // staging loads. Emits global_prefetch_b8 (CDNA5 prefetch path).
  {
    int rp = r + 2;
    if (rp < HW_) {
      const int lines = 5;                       // ceil(66/16) cachelines
      for (int i = tid; i < cin0 * lines; i += NTHREADS) {
        int ci = i / lines, j = i - ci * lines;
        int cc = c0 - 1 + j * 16;
        if (cc < 0) cc = 0;
        if (cc < HW_)
          __builtin_prefetch(&in0[((size_t)(b * cin0 + ci) * HW_ + rp) * HW_ + cc], 0, 3);
      }
    }
  }

  // ---- stage weights for this M-tile: A[m][k], bf16 ----
  for (int i = tid; i < 16 * Kpad; i += NTHREADS) {
    int m = i / Kpad, k = i - m * Kpad;
    float v = 0.f;
    if (k < K) {
      int ci = k / 9, rem = k - ci * 9;
      int mg = mtile * 16 + m;
      if (mg < cout0)      v = w0[(size_t)(mg * cinT + ci) * 9 + rem];
      else if (mg < coutT) v = w1[(size_t)((mg - cout0) * cinT + ci) * 9 + rem];
    }
    sA[m * Kpad + k] = f2bf(v);
  }

  // ---- stage input halo: rows r-1..r+1, cols c0-1..c0+64, all channels ----
  const int OH1 = (up > 0) ? (HW_ / up) : 1;
  for (int i = tid; i < cinT * 3 * 66; i += NTHREADS) {
    int col = i % 66; int t = i / 66;
    int kh = t % 3;   int ci = t / 3;
    int rr = r - 1 + kh;
    int cc = c0 - 1 + col;
    float v = 0.f;
    if (rr >= 0 && rr < HW_ && cc >= 0 && cc < HW_) {
      if (ci < cin0)
        v = in0[((size_t)(b * cin0 + ci) * HW_ + rr) * HW_ + cc];
      else
        v = in1[((size_t)(b * cin1 + (ci - cin0)) * OH1 + rr / up) * OH1 + cc / up];
    }
    sIn[(ci * 3 + kh) * IN_W + col] = f2bf(v);
  }
  __syncthreads();

  // ---- build im2col panel B[n][k] (bf16, K-contiguous for b128 LDS reads) ----
  for (int i = tid; i < PX_PER_BLK * Kpad; i += NTHREADS) {
    int n = i / Kpad, k = i - n * Kpad;
    bf16_t v = f2bf(0.f);
    if (k < K) {
      int ci = k / 9, rem = k - ci * 9, kh = rem / 3, kw = rem - kh * 3;
      v = sIn[(ci * 3 + kh) * IN_W + (n + kw)];
    }
    sB[n * Kpad + k] = v;
  }
  __syncthreads();

  // ---- WMMA main loop ----
  const int wave = tid >> 5, lane = tid & 31;
  const int lm   = lane & 15;          // A row m / D column selector
  const int hi   = (lane >> 4) & 1;    // lane half
  const int nloc = wave * 16 + lm;     // pixel within block
  v8f acc = {};
  const int kbMax = Kpad >> 5;
  for (int kb = 0; kb < kbMax; ++kb) {
    // A 16x32 bf16: lanes 0-15 hold K {0..7, 16..23}, lanes 16-31 {8..15, 24..31}
    const bf16_t* ap = &sA[lm * Kpad + kb * 32 + (hi ? 8 : 0)];
    v8bf a0 = *(const v8bf*)ap;
    v8bf a1 = *(const v8bf*)(ap + 16);
    // B 32x16 bf16: lanes 0-15 hold K {0..7, 16..23}, lanes 16-31 {8..15, 24..31}
    const bf16_t* bp = &sB[nloc * Kpad + kb * 32 + (hi ? 8 : 0)];
    v8bf b0 = *(const v8bf*)bp;
    v8bf b1 = *(const v8bf*)(bp + 16);
    v16bf va, vb;
#pragma unroll
    for (int i = 0; i < 8; ++i) {
      va[i] = a0[i]; va[i + 8] = a1[i];
      vb[i] = b0[i]; vb[i + 8] = b1[i];
    }
    acc = __builtin_amdgcn_wmma_f32_16x16x32_bf16(false, va, false, vb,
                                                  (short)0, acc, false, false);
  }

  // ---- spill D (VGPR j: lanes 0-15 -> M=j, lanes 16-31 -> M=j+8) ----
#pragma unroll
  for (int j = 0; j < 8; ++j)
    sEpi[(j + hi * 8) * PX_PER_BLK + wave * 16 + lm] = acc[j];
  __syncthreads();

  // ---- epilogue ----
  if (!gated) {
    for (int i = tid; i < 16 * PX_PER_BLK; i += NTHREADS) {
      int m = i / PX_PER_BLK, n = i - m * PX_PER_BLK;
      int mg = mtile * 16 + m;
      int col = c0 + n;
      if (mg < coutT && col < HW_) {
        float v = sEpi[i] + bias0[mg];
        out[((size_t)(b * outCstride + chBase + mg) * HW_ + r) * HW_ + col] = v;
      }
    }
  } else {
    for (int i = tid; i < 8 * PX_PER_BLK; i += NTHREADS) {
      int m = i / PX_PER_BLK, n = i - m * PX_PER_BLK;
      int col = c0 + n;
      if (col < HW_) {
        float g  = sEpi[m * PX_PER_BLK + n] + bias0[m];
        float mk = sEpi[(m + 8) * PX_PER_BLK + n] + bias1[m];
        float val = g * (1.f / (1.f + __expf(-mk)));
        out[((size_t)(b * outCstride + m) * HW_ + r) * HW_ + col] = val;
      }
    }
  }
}

// ---------------------------------------------------------------------------
// Exact adaptive average pool (block k), x:[32,8,300,300] -> p:[32,8,300/k,300/k]
// ---------------------------------------------------------------------------
__global__ __launch_bounds__(256)
void DOAM_pool_avg(const float* __restrict__ x, float* __restrict__ p, int k, int OH)
{
  int idx = blockIdx.x * blockDim.x + threadIdx.x;
  int total = BATCH_ * 8 * OH * OH;
  if (idx >= total) return;
  int ox = idx % OH; int t = idx / OH;
  int oy = t % OH;   t /= OH;
  int c  = t % 8;    int b = t / 8;
  float s = 0.f;
  const float* src = x + ((size_t)(b * 8 + c) * HW_ + oy * k) * HW_ + ox * k;
  for (int dy = 0; dy < k; ++dy)
    for (int dx = 0; dx < k; ++dx)
      s += src[(size_t)dy * HW_ + dx];
  p[idx] = s / (float)(k * k);
}

// ---------------------------------------------------------------------------
// Per-channel sum / sum-of-squares over [32,8,300,300]; one block per channel.
// ---------------------------------------------------------------------------
__global__ __launch_bounds__(256)
void DOAM_bn_stats(const float* __restrict__ g, float* __restrict__ stats)
{
  __shared__ float sS[256], sQ[256];
  int c = blockIdx.x;
  float s = 0.f, q = 0.f;
  for (int b = 0; b < BATCH_; ++b) {
    const float* p = g + (size_t)(b * 8 + c) * (HW_ * HW_);
    for (int i = threadIdx.x; i < HW_ * HW_; i += 256) {
      float v = p[i]; s += v; q += v * v;
    }
  }
  sS[threadIdx.x] = s; sQ[threadIdx.x] = q;
  __syncthreads();
  for (int o = 128; o > 0; o >>= 1) {
    if (threadIdx.x < (unsigned)o) {
      sS[threadIdx.x] += sS[threadIdx.x + o];
      sQ[threadIdx.x] += sQ[threadIdx.x + o];
    }
    __syncthreads();
  }
  if (threadIdx.x == 0) { stats[c] = sS[0]; stats[8 + c] = sQ[0]; }
}

// ---------------------------------------------------------------------------
// Fused: BN(gated) -> conv3x3 (8->1) -> sigmoid -> residual gating with gamma.
// One thread per (b, r, c); writes all 3 image channels.
// ---------------------------------------------------------------------------
__global__ __launch_bounds__(256)
void DOAM_attention_out(const float* __restrict__ im, const float* __restrict__ gated,
                        const float* __restrict__ stats,
                        const float* __restrict__ bn_w, const float* __restrict__ bn_b,
                        const float* __restrict__ w11, const float* __restrict__ b11,
                        const float* __restrict__ gamma, float* __restrict__ out)
{
  int idx = blockIdx.x * blockDim.x + threadIdx.x;
  int total = BATCH_ * HW_ * HW_;
  if (idx >= total) return;
  int c = idx % HW_; int t = idx / HW_;
  int r = t % HW_;   int b = t / HW_;

  const float N = (float)BATCH_ * HW_ * HW_;
  float scale[8], shift[8];
#pragma unroll
  for (int ci = 0; ci < 8; ++ci) {
    float mean = stats[ci] / N;
    float var  = stats[8 + ci] / N - mean * mean;
    float inv  = rsqrtf(var + 1e-5f) * bn_w[ci];
    scale[ci] = inv;
    shift[ci] = bn_b[ci] - mean * inv;
  }

  float acc = b11[0];
  for (int ci = 0; ci < 8; ++ci) {
    const float* gp = gated + (size_t)(b * 8 + ci) * (HW_ * HW_);
    const float* wp = w11 + ci * 9;
#pragma unroll
    for (int kh = 0; kh < 3; ++kh) {
      int rr = r + kh - 1;
      if (rr < 0 || rr >= HW_) continue;
#pragma unroll
      for (int kw = 0; kw < 3; ++kw) {
        int cc = c + kw - 1;
        if (cc < 0 || cc >= HW_) continue;
        float v = gp[(size_t)rr * HW_ + cc];
        acc += (v * scale[ci] + shift[ci]) * wp[kh * 3 + kw];
      }
    }
  }
  float att = 1.f / (1.f + __expf(-acc));
  float gm  = gamma[0];
  float wA  = gm * att + (1.f - gm);
#pragma unroll
  for (int ch = 0; ch < 3; ++ch) {
    size_t o = ((size_t)(b * 3 + ch) * HW_ + r) * HW_ + c;
    out[o] = wA * im[o];
  }
}

// ---------------------------------------------------------------------------
extern "C" void kernel_launch(void* const* d_in, const int* in_sizes, int n_in,
                              void* d_out, int out_size, void* d_ws, size_t ws_size,
                              hipStream_t stream)
{
  const float* im  = (const float*)d_in[0];
  const float* w1  = (const float*)d_in[1];  const float* b1  = (const float*)d_in[2];
  const float* w2  = (const float*)d_in[3];  const float* b2  = (const float*)d_in[4];
  const float* w3  = (const float*)d_in[5];  const float* b3  = (const float*)d_in[6];
  const float* w4  = (const float*)d_in[7];  const float* b4  = (const float*)d_in[8];
  const float* w5  = (const float*)d_in[9];  const float* b5  = (const float*)d_in[10];
  const float* wc5 = (const float*)d_in[11]; const float* bc5 = (const float*)d_in[12];
  const float* wc10= (const float*)d_in[13]; const float* bc10= (const float*)d_in[14];
  const float* wc15= (const float*)d_in[15]; const float* bc15= (const float*)d_in[16];
  const float* wg  = (const float*)d_in[17]; const float* bg  = (const float*)d_in[18];
  const float* wm  = (const float*)d_in[19]; const float* bm  = (const float*)d_in[20];
  const float* bnw = (const float*)d_in[21];
  const float* bnb = (const float*)d_in[22];
  const float* w11 = (const float*)d_in[23]; const float* b11 = (const float*)d_in[24];
  const float* gma = (const float*)d_in[25];
  float* out = (float*)d_out;

  const size_t HWp = (size_t)HW_ * HW_;             // 90000
  float* ws   = (float*)d_ws;
  float* bufA = ws;                                  // 32ch: x3, then g_in (24ch)
  float* bufB = bufA + (size_t)BATCH_ * 32 * HWp;    // 16ch: x2, then x4
  float* bufC = bufB + (size_t)BATCH_ * 16 * HWp;    //  8ch: x1, then x5
  float* bufD = bufC + (size_t)BATCH_ * 8  * HWp;    //  8ch: gated (pre-BN)
  float* p5   = bufD + (size_t)BATCH_ * 8  * HWp;    // [32,8,60,60]
  float* p10  = p5   + (size_t)BATCH_ * 8 * 60 * 60; // [32,8,30,30]
  float* p15  = p10  + (size_t)BATCH_ * 8 * 30 * 30; // [32,8,20,20]
  float* stats= p15  + (size_t)BATCH_ * 8 * 20 * 20; // [16]

  const dim3 blk(NTHREADS);
  const int colTiles = (HW_ + PX_PER_BLK - 1) / PX_PER_BLK;  // 5
  auto conv = [&](const float* in0, int cin0, const float* in1, int cin1, int up,
                  const float* ww0, const float* bb0, int cout0,
                  const float* ww1, const float* bb1, int cout1,
                  float* o, int cstride, int chBase, int gated) {
    int mt = (cout0 + cout1 + 15) / 16;
    dim3 grid(colTiles, mt, BATCH_ * HW_);
    DOAM_conv3x3_wmma<<<grid, blk, 0, stream>>>(in0, cin0, in1, cin1, up,
                                                ww0, bb0, cout0, ww1, bb1, cout1,
                                                o, cstride, chBase, gated);
  };

  // conv chain 3->8->16->32->16->8
  conv(im,   3,  nullptr, 0, 0, w1, b1, 8,  nullptr, nullptr, 0, bufC, 8,  0, 0);
  conv(bufC, 8,  nullptr, 0, 0, w2, b2, 16, nullptr, nullptr, 0, bufB, 16, 0, 0);
  conv(bufB, 16, nullptr, 0, 0, w3, b3, 32, nullptr, nullptr, 0, bufA, 32, 0, 0);
  conv(bufA, 32, nullptr, 0, 0, w4, b4, 16, nullptr, nullptr, 0, bufB, 16, 0, 0);
  conv(bufB, 16, nullptr, 0, 0, w5, b5, 8,  nullptr, nullptr, 0, bufC, 8,  0, 0);

  // triple-scale pooled context
  {
    int t5 = BATCH_ * 8 * 60 * 60, t10 = BATCH_ * 8 * 30 * 30, t15 = BATCH_ * 8 * 20 * 20;
    DOAM_pool_avg<<<dim3((t5  + 255) / 256), dim3(256), 0, stream>>>(bufC, p5,  5,  60);
    DOAM_pool_avg<<<dim3((t10 + 255) / 256), dim3(256), 0, stream>>>(bufC, p10, 10, 30);
    DOAM_pool_avg<<<dim3((t15 + 255) / 256), dim3(256), 0, stream>>>(bufC, p15, 15, 20);
  }

  // context convs: concat [x, upsample(pool_k)] handled by two-source gather; g_in = bufA[24ch]
  conv(bufC, 8, p5,  8, 5,  wc5,  bc5,  8, nullptr, nullptr, 0, bufA, 24, 0,  0);
  conv(bufC, 8, p10, 8, 10, wc10, bc10, 8, nullptr, nullptr, 0, bufA, 24, 8,  0);
  conv(bufC, 8, p15, 8, 15, wc15, bc15, 8, nullptr, nullptr, 0, bufA, 24, 16, 0);

  // gated conv: one 16-row GEMM (g rows 0-7 from wg, mask rows 8-15 from wm), fused g*sigmoid(m)
  conv(bufA, 24, nullptr, 0, 0, wg, bg, 8, wm, bm, 8, bufD, 8, 0, 1);

  // batch-stat BN moments
  DOAM_bn_stats<<<dim3(8), dim3(256), 0, stream>>>(bufD, stats);

  // fused BN -> 8->1 conv -> sigmoid -> residual gating
  {
    int total = BATCH_ * HW_ * HW_;
    DOAM_attention_out<<<dim3((total + 255) / 256), dim3(256), 0, stream>>>(
        im, bufD, stats, bnw, bnb, w11, b11, gma, out);
  }
}
